// ViT_52896817218104
// MI455X (gfx1250) — compile-verified, matrix operand order checked by reading, not run on hardware
//
#include <hip/hip_runtime.h>
#include <hip/hip_bf16.h>
#include <stdint.h>

// ---------------------------------------------------------------------------
// ViT-Base forward for MI455X (gfx1250). All matmuls run on v_wmma_f32_16x16x32_f16
// (fp16 in, fp32 accumulate). Compute-bound workload (~0.87 TFLOP, weights fit L2).
// ---------------------------------------------------------------------------

typedef __attribute__((ext_vector_type(16))) _Float16 v16h;
typedef __attribute__((ext_vector_type(8)))  _Float16 h8;
typedef __attribute__((ext_vector_type(8)))  float    v8f;

#define B_      32
#define C_      3
#define IMG_    224
#define PH_     16
#define D_      768
#define H_      12
#define DH_     64
#define LAYERS_ 12
#define FF_     3072
#define NC_     1000
#define SEQ_    197
#define NPATCH_ 196
#define GRIDP_  14          // IMG/PH
#define SPAD_   208         // padded attn row stride (halves), 16B aligned

#define BM 128
#define BN 128
#define BK 64
#define LDS_STRIDE 72       // 64 + 8 halves pad

enum { EPI_NONE = 0, EPI_BIAS = 1, EPI_BIAS_GELU = 2 };

__device__ __forceinline__ float gelu_exact(float v) {
    return 0.5f * v * (1.0f + erff(v * 0.70710678118654752f));
}

// ---------------------------------------------------------------------------
// Generic batched WMMA GEMM.
//   C[z] = scale * A[z] * B[z] (+bias) (+GELU)
// A: (M,K) fp16 row-major, lda.     batch offset = (z/aDiv)*aOuter + (z%aDiv)*aInner
// B: fp16, (K,N) row-major (transB=0) or (N,K) row-major (transB=1), ldb.
// C: fp32 or fp16 (outHalf), ldc.
// Block = 256 threads = 8 waves; tile 128x128; wave tile 32x64; BK=64.
// ---------------------------------------------------------------------------
__global__ __launch_bounds__(256)
void gemm_wmma(const _Float16* __restrict__ A, long aOuter, long aInner, int aDiv, int lda,
               const _Float16* __restrict__ Bm, long bOuter, long bInner, int bDiv, int ldb,
               int transB,
               void* __restrict__ Cp, long cOuter, long cInner, int cDiv, int ldc,
               const float* __restrict__ bias,
               int M, int N, int K, float scale, int epi, int outHalf)
{
    __shared__ __attribute__((aligned(16))) _Float16 As[BM * LDS_STRIDE];
    __shared__ __attribute__((aligned(16))) _Float16 Bs[BN * LDS_STRIDE]; // Bs[n][k]

    const int tid  = threadIdx.x;
    const int lane = tid & 31;
    const int wave = tid >> 5;
    const int wm   = (wave & 3) * 32;   // wave M offset inside tile
    const int wn   = (wave >> 2) * 64;  // wave N offset inside tile
    const int lr   = lane & 15;
    const int hiK  = (lane < 16) ? 0 : 8;   // 16-bit A/B fragment K sub-offset

    const int bm0 = blockIdx.y * BM;
    const int bn0 = blockIdx.x * BN;
    const int z   = blockIdx.z;

    const _Float16* Ab = A  + (long)(z / aDiv) * aOuter + (long)(z % aDiv) * aInner;
    const _Float16* Bb = Bm + (long)(z / bDiv) * bOuter + (long)(z % bDiv) * bInner;
    const long cOff    =     (long)(z / cDiv) * cOuter + (long)(z % cDiv) * cInner;

    v8f acc[2][4];
#pragma unroll
    for (int f = 0; f < 2; ++f)
#pragma unroll
        for (int g = 0; g < 4; ++g) acc[f][g] = (v8f){0.f,0.f,0.f,0.f,0.f,0.f,0.f,0.f};

    for (int kt = 0; kt < K; kt += BK) {
        // ---- stage A tile: As[m][k], 1024 chunks of 8 halves ----
#pragma unroll
        for (int it = 0; it < 4; ++it) {
            int c  = it * 256 + tid;
            int m  = c >> 3;
            int kc = (c & 7) << 3;
            int gm = bm0 + m;
            int gk = kt + kc;
            h8 val = {};
            if (gm < M) {
                const _Float16* src = Ab + (long)gm * lda + gk;
                if (gk + 8 <= K && ((((uintptr_t)src) & 15) == 0)) {
                    val = *(const h8*)src;
                } else {
#pragma unroll
                    for (int j = 0; j < 8; ++j) if (gk + j < K) val[j] = src[j];
                }
            }
            *(h8*)&As[m * LDS_STRIDE + kc] = val;
        }
        // ---- stage B tile as Bs[n][k] ----
        if (transB) {
#pragma unroll
            for (int it = 0; it < 4; ++it) {
                int c  = it * 256 + tid;
                int n  = c >> 3;
                int kc = (c & 7) << 3;
                int gn = bn0 + n;
                int gk = kt + kc;
                h8 val = {};
                if (gn < N) {
                    const _Float16* src = Bb + (long)gn * ldb + gk;
                    if (gk + 8 <= K && ((((uintptr_t)src) & 15) == 0)) {
                        val = *(const h8*)src;
                    } else {
#pragma unroll
                        for (int j = 0; j < 8; ++j) if (gk + j < K) val[j] = src[j];
                    }
                }
                *(h8*)&Bs[n * LDS_STRIDE + kc] = val;
            }
        } else {
#pragma unroll
            for (int it = 0; it < 4; ++it) {
                int c  = it * 256 + tid;
                int k  = c >> 4;            // 0..63
                int nc = (c & 15) << 3;     // 0..120
                int gk = kt + k;
                int gn = bn0 + nc;
                h8 val = {};
                if (gk < K) {
                    const _Float16* src = Bb + (long)gk * ldb + gn;
                    if (gn + 8 <= N && ((((uintptr_t)src) & 15) == 0)) {
                        val = *(const h8*)src;
                    } else {
#pragma unroll
                        for (int j = 0; j < 8; ++j) if (gn + j < N) val[j] = src[j];
                    }
                }
#pragma unroll
                for (int j = 0; j < 8; ++j) Bs[(nc + j) * LDS_STRIDE + k] = val[j];
            }
        }
        __syncthreads();

        // speculative prefetch of next A tile (global_prefetch_b8)
        if (kt + BK < K && bm0 + (tid >> 1) < M)
            __builtin_prefetch(Ab + (long)(bm0 + (tid >> 1)) * lda + kt + BK, 0, 1);

        // ---- 2 x 8 WMMAs over this K slab ----
#pragma unroll
        for (int kk = 0; kk < BK; kk += 32) {
            v16h afr[2], bfr[4];
#pragma unroll
            for (int f = 0; f < 2; ++f) {
                const _Float16* p = &As[(wm + f * 16 + lr) * LDS_STRIDE + kk + hiK];
                h8 lo = *(const h8*)p;
                h8 hi = *(const h8*)(p + 16);
                afr[f] = __builtin_shufflevector(lo, hi, 0,1,2,3,4,5,6,7,8,9,10,11,12,13,14,15);
            }
#pragma unroll
            for (int g = 0; g < 4; ++g) {
                const _Float16* p = &Bs[(wn + g * 16 + lr) * LDS_STRIDE + kk + hiK];
                h8 lo = *(const h8*)p;
                h8 hi = *(const h8*)(p + 16);
                bfr[g] = __builtin_shufflevector(lo, hi, 0,1,2,3,4,5,6,7,8,9,10,11,12,13,14,15);
            }
#pragma unroll
            for (int f = 0; f < 2; ++f)
#pragma unroll
                for (int g = 0; g < 4; ++g)
                    acc[f][g] = __builtin_amdgcn_wmma_f32_16x16x32_f16(
                        false, afr[f], false, bfr[g], (short)0, acc[f][g], false, false);
        }
        __syncthreads();
    }

    // ---- epilogue: scale + bias + optional GELU, fp32 or fp16 store ----
    float*    Cf = (float*)Cp    + cOff;
    _Float16* Ch = (_Float16*)Cp + cOff;
#pragma unroll
    for (int f = 0; f < 2; ++f) {
#pragma unroll
        for (int g = 0; g < 4; ++g) {
            int n = bn0 + wn + g * 16 + lr;
            if (n >= N) continue;
            float bval = (epi != EPI_NONE && bias) ? bias[n] : 0.f;
#pragma unroll
            for (int i = 0; i < 8; ++i) {
                int m = bm0 + wm + f * 16 + ((lane < 16) ? i : i + 8);
                if (m >= M) continue;
                float v = acc[f][g][i] * scale + bval;
                if (epi == EPI_BIAS_GELU) v = gelu_exact(v);
                if (outHalf) Ch[(long)m * ldc + n] = (_Float16)v;
                else         Cf[(long)m * ldc + n] = v;
            }
        }
    }
}

// ---------------------------------------------------------------------------
// LayerNorm: one block per row, fp32 in, fp16 out (feeds a GEMM).
// ---------------------------------------------------------------------------
__global__ __launch_bounds__(256)
void layernorm_h(const float* __restrict__ x, const float* __restrict__ g,
                 const float* __restrict__ b, _Float16* __restrict__ out,
                 int cols, float eps)
{
    __shared__ float red[256];
    const long row = blockIdx.x;
    const float* xr = x + row * cols;
    const int tid = threadIdx.x;

    float s = 0.f;
    for (int j = tid; j < cols; j += 256) s += xr[j];
    red[tid] = s; __syncthreads();
    for (int st = 128; st > 0; st >>= 1) { if (tid < st) red[tid] += red[tid + st]; __syncthreads(); }
    const float mean = red[0] / cols; __syncthreads();

    float v = 0.f;
    for (int j = tid; j < cols; j += 256) { float d = xr[j] - mean; v += d * d; }
    red[tid] = v; __syncthreads();
    for (int st = 128; st > 0; st >>= 1) { if (tid < st) red[tid] += red[tid + st]; __syncthreads(); }
    const float rstd = rsqrtf(red[0] / cols + eps);

    _Float16* orow = out + row * cols;
    for (int j = tid; j < cols; j += 256)
        orow[j] = (_Float16)((xr[j] - mean) * rstd * g[j] + b[j]);
}

// ---------------------------------------------------------------------------
// Softmax over SEQ keys; applies 1/sqrt(DH) scale; writes fp16 rows padded
// to SPAD for aligned WMMA fragment loads.
// ---------------------------------------------------------------------------
__global__ __launch_bounds__(256)
void softmax_h(const float* __restrict__ sc, _Float16* __restrict__ attn, float scale)
{
    __shared__ float red[256];
    const long row = blockIdx.x;               // 0 .. B*H*SEQ-1
    const float* sr = sc + row * SEQ_;
    _Float16* ar = attn + row * SPAD_;
    const int tid = threadIdx.x;

    float mx = -3.4e38f;
    for (int j = tid; j < SEQ_; j += 256) mx = fmaxf(mx, sr[j] * scale);
    red[tid] = mx; __syncthreads();
    for (int st = 128; st > 0; st >>= 1) { if (tid < st) red[tid] = fmaxf(red[tid], red[tid + st]); __syncthreads(); }
    mx = red[0]; __syncthreads();

    float s = 0.f;
    for (int j = tid; j < SEQ_; j += 256) s += __expf(sr[j] * scale - mx);
    red[tid] = s; __syncthreads();
    for (int st = 128; st > 0; st >>= 1) { if (tid < st) red[tid] += red[tid + st]; __syncthreads(); }
    const float inv = 1.f / red[0];

    for (int j = tid; j < SEQ_; j += 256)
        ar[j] = (_Float16)(__expf(sr[j] * scale - mx) * inv);
}

// ---------------------------------------------------------------------------
// Elementwise helpers
// ---------------------------------------------------------------------------
__global__ void f32_to_f16(const float* __restrict__ in, _Float16* __restrict__ out, long n) {
    long i = (long)blockIdx.x * 256 + threadIdx.x;
    long stride = (long)gridDim.x * 256;
    for (; i < n; i += stride) out[i] = (_Float16)in[i];
}

__global__ void add_f32(const float* __restrict__ a, const float* __restrict__ b,
                        float* __restrict__ o, long n) {
    long i = (long)blockIdx.x * 256 + threadIdx.x;
    if (i < n) o[i] = a[i] + b[i];
}

// images (B,C,224,224) -> fp16 patch matrix (B*196, 768)
__global__ void patchify(const float* __restrict__ img, _Float16* __restrict__ p) {
    long i = (long)blockIdx.x * 256 + threadIdx.x;
    const long total = (long)B_ * NPATCH_ * D_;
    if (i >= total) return;
    int j  = (int)(i % D_);
    long t = i / D_;
    int n  = (int)(t % NPATCH_);
    int b  = (int)(t / NPATCH_);
    int c  = j >> 8;                  // / 256
    int r  = j & 255;
    int ph = r >> 4, pw = r & 15;
    int hp = n / GRIDP_, wp = n % GRIDP_;
    int row = hp * PH_ + ph, col = wp * PH_ + pw;
    p[i] = (_Float16)img[(((long)b * C_ + c) * IMG_ + row) * IMG_ + col];
}

// x[b,0,:] = cls + pos[0];  x[b,1+n,:] = pe[b,n,:] + bp + pos[1+n]
__global__ void assemble_embed(const float* __restrict__ pe, const float* __restrict__ cls,
                               const float* __restrict__ pos, const float* __restrict__ bp,
                               float* __restrict__ x) {
    long i = (long)blockIdx.x * 256 + threadIdx.x;
    const long total = (long)B_ * SEQ_ * D_;
    if (i >= total) return;
    int d  = (int)(i % D_);
    long t = i / D_;
    int s  = (int)(t % SEQ_);
    int b  = (int)(t / SEQ_);
    float v = (s == 0) ? cls[d]
                       : pe[((long)b * NPATCH_ + (s - 1)) * D_ + d] + bp[d];
    x[i] = v + pos[(long)s * D_ + d];
}

__global__ void extract_cls(const float* __restrict__ x, float* __restrict__ cls) {
    int i = blockIdx.x * 256 + threadIdx.x;
    if (i >= B_ * D_) return;
    int b = i / D_, d = i % D_;
    cls[i] = x[(long)b * SEQ_ * D_ + d];
}

// ---------------------------------------------------------------------------
// Host orchestration
// ---------------------------------------------------------------------------
extern "C" void kernel_launch(void* const* d_in, const int* in_sizes, int n_in,
                              void* d_out, int out_size, void* d_ws, size_t ws_size,
                              hipStream_t stream)
{
    (void)in_sizes; (void)n_in; (void)out_size; (void)ws_size;

    const float* images = (const float*)d_in[0];
    const float* cls_e  = (const float*)d_in[1];
    const float* pos_e  = (const float*)d_in[2];
    const float* Wp     = (const float*)d_in[3];
    const float* bp     = (const float*)d_in[4];
    const float* ln1g   = (const float*)d_in[5];
    const float* ln1b   = (const float*)d_in[6];
    const float* ln2g   = (const float*)d_in[7];
    const float* ln2b   = (const float*)d_in[8];
    const float* Wq     = (const float*)d_in[9];
    const float* bq     = (const float*)d_in[10];
    const float* Wk     = (const float*)d_in[11];
    const float* bk     = (const float*)d_in[12];
    const float* Wv     = (const float*)d_in[13];
    const float* bv     = (const float*)d_in[14];
    const float* Wo     = (const float*)d_in[15];
    const float* bo     = (const float*)d_in[16];
    const float* W1     = (const float*)d_in[17];
    const float* b1     = (const float*)d_in[18];
    const float* W2     = (const float*)d_in[19];
    const float* b2     = (const float*)d_in[20];
    const float* lncg   = (const float*)d_in[21];
    const float* lncb   = (const float*)d_in[22];
    const float* Wc1    = (const float*)d_in[23];
    const float* bc1    = (const float*)d_in[24];
    const float* Wc2    = (const float*)d_in[25];
    const float* bc2    = (const float*)d_in[26];

    char* wsp = (char*)d_ws;
    size_t off = 0;
    auto alloc = [&](size_t bytes) -> void* {
        void* p = wsp + off;
        off = (off + bytes + 255) & ~(size_t)255;
        return p;
    };

    // fp16 weight mirrors (live in L2: ~143 MB)
    _Float16* wp_h  = (_Float16*)alloc(sizeof(_Float16) * (size_t)D_ * D_);
    _Float16* wq_h  = (_Float16*)alloc(sizeof(_Float16) * (size_t)LAYERS_ * DH_ * DH_);
    _Float16* wk_h  = (_Float16*)alloc(sizeof(_Float16) * (size_t)LAYERS_ * DH_ * DH_);
    _Float16* wv_h  = (_Float16*)alloc(sizeof(_Float16) * (size_t)LAYERS_ * DH_ * DH_);
    _Float16* wo_h  = (_Float16*)alloc(sizeof(_Float16) * (size_t)LAYERS_ * D_ * D_);
    _Float16* w1_h  = (_Float16*)alloc(sizeof(_Float16) * (size_t)LAYERS_ * D_ * FF_);
    _Float16* w2_h  = (_Float16*)alloc(sizeof(_Float16) * (size_t)LAYERS_ * FF_ * D_);
    _Float16* wc1_h = (_Float16*)alloc(sizeof(_Float16) * (size_t)D_ * 4 * NC_);
    _Float16* wc2_h = (_Float16*)alloc(sizeof(_Float16) * (size_t)4 * NC_ * NC_);

    // activations
    const long TOK  = (long)B_ * SEQ_;           // 6304 token rows
    const long NXD  = TOK * D_;
    _Float16* p_h    = (_Float16*)alloc(sizeof(_Float16) * (size_t)B_ * NPATCH_ * D_);
    float*    pe_tmp = (float*)   alloc(sizeof(float)    * (size_t)B_ * NPATCH_ * D_);
    float*    x      = (float*)   alloc(sizeof(float)    * (size_t)NXD);
    float*    addbuf = (float*)   alloc(sizeof(float)    * (size_t)NXD);
    float*    tmp_f  = (float*)   alloc(sizeof(float)    * (size_t)NXD);
    _Float16* n_h    = (_Float16*)alloc(sizeof(_Float16) * (size_t)NXD);
    _Float16* q_h    = (_Float16*)alloc(sizeof(_Float16) * (size_t)NXD);
    _Float16* k_h    = (_Float16*)alloc(sizeof(_Float16) * (size_t)NXD);
    _Float16* v_h    = (_Float16*)alloc(sizeof(_Float16) * (size_t)NXD);
    float*    scores = (float*)   alloc(sizeof(float)    * (size_t)B_ * H_ * SEQ_ * SEQ_);
    _Float16* attn_h = (_Float16*)alloc(sizeof(_Float16) * (size_t)B_ * H_ * SEQ_ * SPAD_);
    _Float16* o_h    = (_Float16*)alloc(sizeof(_Float16) * (size_t)NXD);
    _Float16* ff_h   = (_Float16*)alloc(sizeof(_Float16) * (size_t)TOK * FF_);
    float*    cls_f  = (float*)   alloc(sizeof(float)    * (size_t)B_ * D_);
    _Float16* clsn_h = (_Float16*)alloc(sizeof(_Float16) * (size_t)B_ * D_);
    _Float16* h1_h   = (_Float16*)alloc(sizeof(_Float16) * (size_t)B_ * 4 * NC_);

    auto cvt = [&](const float* src, _Float16* dst, long n) {
        int blocks = (int)((n + 255) / 256);
        if (blocks > 32768) blocks = 32768;
        f32_to_f16<<<blocks, 256, 0, stream>>>(src, dst, n);
    };
    auto gemm = [&](const _Float16* A, long aO, long aI, int aDiv, int lda,
                    const _Float16* Bm, long bO, long bI, int bDiv, int ldb, int transB,
                    void* C, long cO, long cI, int cDiv, int ldc,
                    const float* bias, int M, int N, int K,
                    float scale, int epi, int outHalf, int batch) {
        dim3 grid((N + BN - 1) / BN, (M + BM - 1) / BM, batch);
        gemm_wmma<<<grid, 256, 0, stream>>>(A, aO, aI, aDiv, lda,
                                            Bm, bO, bI, bDiv, ldb, transB,
                                            C, cO, cI, cDiv, ldc,
                                            bias, M, N, K, scale, epi, outHalf);
    };

    // ---- weight conversion (deterministic each call) ----
    cvt(Wp,  wp_h,  (long)D_ * D_);
    cvt(Wq,  wq_h,  (long)LAYERS_ * DH_ * DH_);
    cvt(Wk,  wk_h,  (long)LAYERS_ * DH_ * DH_);
    cvt(Wv,  wv_h,  (long)LAYERS_ * DH_ * DH_);
    cvt(Wo,  wo_h,  (long)LAYERS_ * D_ * D_);
    cvt(W1,  w1_h,  (long)LAYERS_ * D_ * FF_);
    cvt(W2,  w2_h,  (long)LAYERS_ * FF_ * D_);
    cvt(Wc1, wc1_h, (long)D_ * 4 * NC_);
    cvt(Wc2, wc2_h, (long)4 * NC_ * NC_);

    // ---- patch embedding ----
    {
        long n = (long)B_ * NPATCH_ * D_;
        patchify<<<(int)((n + 255) / 256), 256, 0, stream>>>(images, p_h);
        gemm(p_h, 0, 0, 1, D_,  wp_h, 0, 0, 1, D_, 0,
             pe_tmp, 0, 0, 1, D_, nullptr,
             B_ * NPATCH_, D_, D_, 1.f, EPI_NONE, 0, 1);
        long nx = NXD;
        assemble_embed<<<(int)((nx + 255) / 256), 256, 0, stream>>>(pe_tmp, cls_e, pos_e, bp, x);
    }

    const int BH = B_ * H_;
    const long SD  = (long)SEQ_ * DH_;     // per-(b,h) q/k/v matrix
    const long SS  = (long)SEQ_ * SEQ_;    // per-(b,h) score matrix
    const long SP  = (long)SEQ_ * SPAD_;   // per-(b,h) padded attn matrix
    const long SDr = (long)SEQ_ * D_;      // per-b token block

    for (int l = 0; l < LAYERS_; ++l) {
        const _Float16* wq = wq_h + (long)l * DH_ * DH_;
        const _Float16* wk = wk_h + (long)l * DH_ * DH_;
        const _Float16* wv = wv_h + (long)l * DH_ * DH_;
        const _Float16* wo = wo_h + (long)l * D_ * D_;
        const _Float16* w1 = w1_h + (long)l * D_ * FF_;
        const _Float16* w2 = w2_h + (long)l * FF_ * D_;

        // LN1 -> fp16
        layernorm_h<<<(int)TOK, 256, 0, stream>>>(x, ln1g + (long)l * D_, ln1b + (long)l * D_,
                                                  n_h, D_, 1e-6f);

        // per-head QKV projections (batched over B*H): (197x64)x(64x64)+bias -> fp16
        gemm(n_h, SDr, DH_, H_, D_,  wq, 0, 0, 1, DH_, 0,
             q_h, SD, 0, 1, DH_, bq + (long)l * DH_, SEQ_, DH_, DH_, 1.f, EPI_BIAS, 1, BH);
        gemm(n_h, SDr, DH_, H_, D_,  wk, 0, 0, 1, DH_, 0,
             k_h, SD, 0, 1, DH_, bk + (long)l * DH_, SEQ_, DH_, DH_, 1.f, EPI_BIAS, 1, BH);
        gemm(n_h, SDr, DH_, H_, D_,  wv, 0, 0, 1, DH_, 0,
             v_h, SD, 0, 1, DH_, bv + (long)l * DH_, SEQ_, DH_, DH_, 1.f, EPI_BIAS, 1, BH);

        // scores = q @ k^T  (batched, transB) -> fp32
        gemm(q_h, SD, 0, 1, DH_,  k_h, SD, 0, 1, DH_, 1,
             scores, SS, 0, 1, SEQ_, nullptr, SEQ_, SEQ_, DH_, 1.f, EPI_NONE, 0, BH);

        // softmax (scale 1/8) -> padded fp16 attn
        softmax_h<<<BH * SEQ_, 256, 0, stream>>>(scores, attn_h, 0.125f);

        // o = attn @ v, scattered back into (B,SEQ,D) head-interleaved fp16
        gemm(attn_h, SP, 0, 1, SPAD_,  v_h, SD, 0, 1, DH_, 0,
             o_h, SDr, DH_, H_, D_, nullptr, SEQ_, DH_, SEQ_, 1.f, EPI_NONE, 1, BH);

        // Wo projection + bias -> fp32, then residual add
        gemm(o_h, 0, 0, 1, D_,  wo, 0, 0, 1, D_, 0,
             tmp_f, 0, 0, 1, D_, bo + (long)l * D_, (int)TOK, D_, D_, 1.f, EPI_BIAS, 0, 1);
        add_f32<<<(int)((NXD + 255) / 256), 256, 0, stream>>>(x, tmp_f, addbuf, NXD);

        // LN2 -> fp16
        layernorm_h<<<(int)TOK, 256, 0, stream>>>(addbuf, ln2g + (long)l * D_, ln2b + (long)l * D_,
                                                  n_h, D_, 1e-6f);

        // FF1 + bias + GELU -> fp16
        gemm(n_h, 0, 0, 1, D_,  w1, 0, 0, 1, FF_, 0,
             ff_h, 0, 0, 1, FF_, b1 + (long)l * FF_, (int)TOK, FF_, D_, 1.f, EPI_BIAS_GELU, 1, 1);

        // FF2 + bias -> fp32, residual -> next x
        gemm(ff_h, 0, 0, 1, FF_,  w2, 0, 0, 1, D_, 0,
             tmp_f, 0, 0, 1, D_, b2 + (long)l * D_, (int)TOK, D_, FF_, 1.f, EPI_BIAS, 0, 1);
        add_f32<<<(int)((NXD + 255) / 256), 256, 0, stream>>>(tmp_f, addbuf, x, NXD);
    }

    // ---- classifier head ----
    extract_cls<<<(B_ * D_ + 255) / 256, 256, 0, stream>>>(x, cls_f);
    layernorm_h<<<B_, 256, 0, stream>>>(cls_f, lncg, lncb, clsn_h, D_, 1e-5f);
    gemm(clsn_h, 0, 0, 1, D_,  wc1_h, 0, 0, 1, 4 * NC_, 0,
         h1_h, 0, 0, 1, 4 * NC_, bc1, B_, 4 * NC_, D_, 1.f, EPI_BIAS_GELU, 1, 1);
    gemm(h1_h, 0, 0, 1, 4 * NC_,  wc2_h, 0, 0, 1, NC_, 0,
         d_out, 0, 0, 1, NC_, bc2, B_, NC_, 4 * NC_, 1.f, EPI_BIAS, 0, 1);
}
